// EncoderLayer_68186900791439
// MI455X (gfx1250) — matrix operand order repaired and checked by reading, not compiled, not run on hardware
//
#include <hip/hip_runtime.h>
#include <math.h>

typedef __attribute__((ext_vector_type(16))) __bf16 v16bf;
typedef __attribute__((ext_vector_type(8)))  __bf16 v8bf;
typedef __attribute__((ext_vector_type(8)))  float  v8f;

#define LDSW 136   // 128 + 8 bf16 pad: keeps 16B alignment, staggers banks

__device__ inline unsigned short bfbits(float f) {
    unsigned u = __builtin_bit_cast(unsigned, f);
    unsigned r = u + 0x7fffu + ((u >> 16) & 1u);   // round-to-nearest-even
    return (unsigned short)(r >> 16);
}
__device__ inline __bf16 f2bf(float f) { return __builtin_bit_cast(__bf16, bfbits(f)); }

__device__ inline v8f wmma_bf16(v16bf a, v16bf b, v8f c) {
    return __builtin_amdgcn_wmma_f32_16x16x32_bf16(false, a, false, b, (short)0, c, false, false);
}

// A-fragment (contiguous K order, matching packed B): lanes 0-15 hold K=0..15,
// lanes 16-31 hold K=16..31; ss = (lane>=16)?8:0.
__device__ inline v16bf load_afrag(const __bf16* ly, int mr, int kt, int ss) {
    const __bf16* rp = ly + mr * LDSW + kt * 32 + 2 * ss;
    v8bf lo = *(const v8bf*)(rp);
    v8bf hi = *(const v8bf*)(rp + 8);
    return __builtin_shufflevector(lo, hi, 0,1,2,3,4,5,6,7,8,9,10,11,12,13,14,15);
}

__device__ inline void load_bfrags(const __bf16* bp, v16bf bf4[4]) {
#pragma unroll
    for (int kt = 0; kt < 4; ++kt) bf4[kt] = *(const v16bf*)(bp + kt * 512);
}

// LayerNorm 128 rows -> bf16 LDS. 2 threads per row, pair-reduce via shfl_xor(1).
__device__ inline void ln_to_lds(const float* __restrict__ xsrc, const float* __restrict__ gam,
                                 const float* __restrict__ bet, __bf16* ly, int row0, int Nn) {
    int tid = threadIdx.x;
    int r = tid >> 1, half = tid & 1;
    int grow = row0 + r;
    float4 xr[16];
    float s = 0.f, sq = 0.f;
    if (grow < Nn) {
        const float4* xp = (const float4*)(xsrc + (size_t)grow * 128 + half * 64);
#pragma unroll
        for (int i = 0; i < 16; ++i) {
            float4 v = xp[i]; xr[i] = v;
            s  += v.x + v.y + v.z + v.w;
            sq += v.x*v.x + v.y*v.y + v.z*v.z + v.w*v.w;
        }
    } else {
#pragma unroll
        for (int i = 0; i < 16; ++i) xr[i] = make_float4(0.f, 0.f, 0.f, 0.f);
    }
    s  += __shfl_xor(s, 1);
    sq += __shfl_xor(sq, 1);
    float mu   = s * (1.f / 128.f);
    float var  = sq * (1.f / 128.f) - mu * mu;
    float rinv = 1.0f / sqrtf(var + 1e-5f);
#pragma unroll
    for (int i = 0; i < 16; ++i) {
        int c = half * 64 + i * 4;
        float4 v = xr[i];
        float y0 = (v.x - mu) * rinv * gam[c+0] + bet[c+0];
        float y1 = (v.y - mu) * rinv * gam[c+1] + bet[c+1];
        float y2 = (v.z - mu) * rinv * gam[c+2] + bet[c+2];
        float y3 = (v.w - mu) * rinv * gam[c+3] + bet[c+3];
        uint2 pk;
        pk.x = (unsigned)bfbits(y0) | ((unsigned)bfbits(y1) << 16);
        pk.y = (unsigned)bfbits(y2) | ((unsigned)bfbits(y3) << 16);
        *(uint2*)(ly + r * LDSW + c) = pk;
    }
}

// ---- kernel 1: repack weights into WMMA B-operand tile order (bf16) ----
// Packed B order (contiguous K, this code's choice, not AMD's table order): lane = col n (lane&15); K = (lane>=16?16:0)+e, e=0..15.
// tile(nt,kt) stored contiguously: elem index ((nt*4+kt)*32 + lane)*16 + e.
__global__ void k_pack(const float* __restrict__ Wq, const float* __restrict__ Wk,
                       const float* __restrict__ Wv, const float* __restrict__ Wsk,
                       const float* __restrict__ W1, const float* __restrict__ W2,
                       const float* __restrict__ bq, const float* __restrict__ bk,
                       const float* __restrict__ bv, const float* __restrict__ bsk,
                       __bf16* __restrict__ pw0, __bf16* __restrict__ pw1,
                       __bf16* __restrict__ pw2, float* __restrict__ bias0) {
    int t = blockIdx.x * 256 + threadIdx.x;
    if (t < 114688) {                       // 56 n-tiles x 4 k-tiles of [Wq|Wk|Wv|Wskip]
        int e = t & 15, lane = (t >> 4) & 31, kt = (t >> 9) & 3, nt = t >> 11;
        int n = nt * 16 + (lane & 15);
        int k = kt * 32 + ((lane >> 4) << 4) + e;
        float val;
        if      (n < 256) val = Wq [k * 256 +  n       ];
        else if (n < 512) val = Wk [k * 256 + (n - 256)];
        else if (n < 768) val = Wv [k * 256 + (n - 512)];
        else              val = Wsk[k * 128 + (n - 768)];
        pw0[t] = f2bf(val);
    } else if (t < 114688 + 16384) {        // W1: 8 x 4 tiles
        int u = t - 114688;
        int e = u & 15, lane = (u >> 4) & 31, kt = (u >> 9) & 3, nt = u >> 11;
        int n = nt * 16 + (lane & 15);
        int k = kt * 32 + ((lane >> 4) << 4) + e;
        pw1[u] = f2bf(W1[k * 128 + n]);
    } else if (t < 114688 + 32768) {        // W2
        int u = t - 114688 - 16384;
        int e = u & 15, lane = (u >> 4) & 31, kt = (u >> 9) & 3, nt = u >> 11;
        int n = nt * 16 + (lane & 15);
        int k = kt * 32 + ((lane >> 4) << 4) + e;
        pw2[u] = f2bf(W2[k * 128 + n]);
    } else if (t < 114688 + 32768 + 896) {  // fused bias [bq|bk|bv|bskip]
        int i = t - 114688 - 32768;
        float v;
        if      (i < 256) v = bq [i      ];
        else if (i < 512) v = bk [i - 256];
        else if (i < 768) v = bv [i - 512];
        else              v = bsk[i - 768];
        bias0[i] = v;
    }
}

// ---- kernel 2: init segment-softmax state ----
__global__ void k_init(float* __restrict__ attn, float* __restrict__ denom,
                       float* __restrict__ mbuf, int Nn) {
    int t = blockIdx.x * 256 + threadIdx.x;
    int n256 = Nn * 256, n2 = Nn * 2;
    if      (t < n256)           attn [t]              = 0.f;
    else if (t < n256 + n2)      denom[t - n256]       = 0.f;
    else if (t < n256 + 2 * n2)  mbuf [t - n256 - n2]  = -3.402823466e38f;
}

// ---- kernel 3: LN1 + fused [Q|K|V|skip] GEMM (N x 128) @ (128 x 896), bf16 WMMA ----
// Double-buffered B-fragments: loads for tile nt+1 issue before WMMAs of tile nt,
// so s_wait_loadcnt sinks past the matrix ops and load latency is hidden.
__global__ __launch_bounds__(256) void k_ln1_qkvs(
    const float* __restrict__ x, const float* __restrict__ g1, const float* __restrict__ b1v,
    const __bf16* __restrict__ pw0, const float* __restrict__ bias0,
    float* __restrict__ qkvs, int Nn) {
    __shared__ __bf16 ly[128 * LDSW];
    int row0 = blockIdx.x * 128;
    ln_to_lds(x, g1, b1v, ly, row0, Nn);
    __syncthreads();

    int wave = threadIdx.x >> 5, lane = threadIdx.x & 31;
    int m0 = wave * 16;
    int mr = m0 + (lane & 15);
    int ss = (lane >> 4) * 8;
    v16bf af[4];
#pragma unroll
    for (int kt = 0; kt < 4; ++kt) af[kt] = load_afrag(ly, mr, kt, ss);

    int nlane = lane & 15;
    int rbase = row0 + m0 + ((lane >> 4) << 3);   // C/D: lanes 16-31 are rows +8
    bool full = (row0 + 128) <= Nn;               // block-uniform fast path

    const __bf16* bpbase = pw0 + lane * 16;
    v16bf bcur[4], bnxt[4];
    load_bfrags(bpbase, bcur);
    for (int nt = 0; nt < 56; ++nt) {
        if (nt + 1 < 56) load_bfrags(bpbase + (size_t)(nt + 1) * 2048, bnxt);
        v8f acc = {0.f, 0.f, 0.f, 0.f, 0.f, 0.f, 0.f, 0.f};
#pragma unroll
        for (int kt = 0; kt < 4; ++kt) acc = wmma_bf16(af[kt], bcur[kt], acc);
        int n = nt * 16 + nlane;
        float bs = bias0[n];
        if (full) {
#pragma unroll
            for (int rr = 0; rr < 8; ++rr)
                qkvs[(size_t)(rbase + rr) * 896 + n] = acc[rr] + bs;
        } else {
#pragma unroll
            for (int rr = 0; rr < 8; ++rr) {
                int gr = rbase + rr;
                if (gr < Nn) qkvs[(size_t)gr * 896 + n] = acc[rr] + bs;
            }
        }
#pragma unroll
        for (int kt = 0; kt < 4; ++kt) bcur[kt] = bnxt[kt];
    }
}

// ---- kernel 4: per-(edge,head) attention logits + segment max ----
__device__ inline void atomicMaxF(float* addr, float val) {
    if (val >= 0.f) atomicMax((int*)addr, __float_as_int(val));
    else            atomicMin((unsigned int*)addr, (unsigned int)__float_as_int(val));
}

__global__ void k_edge_logits(const long long* __restrict__ ei, const float* __restrict__ qkvs,
                              float* __restrict__ alpha, float* __restrict__ mbuf, int Ee) {
    int t = blockIdx.x * 256 + threadIdx.x;
    if (t >= 2 * Ee) return;
    int e = t >> 1, h = t & 1;
    long long s = ei[e], d = ei[Ee + e];
    const float4* qp = (const float4*)(qkvs + (size_t)d * 896 +       h * 128);
    const float4* kp = (const float4*)(qkvs + (size_t)s * 896 + 256 + h * 128);
    float acc = 0.f;
#pragma unroll
    for (int i = 0; i < 32; ++i) {
        float4 q4 = qp[i], k4 = kp[i];
        acc += q4.x*k4.x + q4.y*k4.y + q4.z*k4.z + q4.w*k4.w;
    }
    float a = acc * 0.0883883476483184f;   // 1/sqrt(128)
    alpha[t] = a;
    atomicMaxF(&mbuf[d * 2 + h], a);
}

// ---- kernel 5: softmax numerators, denom + weighted-V scatter (1 wave / edge) ----
__global__ void k_edge_accum(const long long* __restrict__ ei, const float* __restrict__ qkvs,
                             const float* __restrict__ alpha, const float* __restrict__ mbuf,
                             float* __restrict__ denom, float* __restrict__ attn, int Ee) {
    int wid = blockIdx.x * 8 + (threadIdx.x >> 5);
    if (wid >= Ee) return;
    int lane = threadIdx.x & 31;
    long long s = ei[wid], d = ei[Ee + wid];
    float p0 = expf(alpha[2 * wid]     - mbuf[d * 2]);
    float p1 = expf(alpha[2 * wid + 1] - mbuf[d * 2 + 1]);
    if (lane == 0) atomicAdd(&denom[d * 2],     p0);
    if (lane == 1) atomicAdd(&denom[d * 2 + 1], p1);
    const float* vb = qkvs + (size_t)s * 896 + 512;
    float4 v0 = *(const float4*)(vb +       lane * 4);
    float4 v1 = *(const float4*)(vb + 128 + lane * 4);
    float* a0 = attn + (size_t)d * 256 +       lane * 4;
    atomicAdd(a0 + 0, p0 * v0.x); atomicAdd(a0 + 1, p0 * v0.y);
    atomicAdd(a0 + 2, p0 * v0.z); atomicAdd(a0 + 3, p0 * v0.w);
    float* a1 = attn + (size_t)d * 256 + 128 + lane * 4;
    atomicAdd(a1 + 0, p1 * v1.x); atomicAdd(a1 + 1, p1 * v1.y);
    atomicAdd(a1 + 2, p1 * v1.z); atomicAdd(a1 + 3, p1 * v1.w);
}

// ---- kernel 6: normalize, head-mean, + skip + residual ----
__global__ void k_combine(const float* __restrict__ x, const float* __restrict__ qkvs,
                          const float* __restrict__ attn, const float* __restrict__ denom,
                          float* __restrict__ x1, int Nn) {
    int t = blockIdx.x * 256 + threadIdx.x;
    if (t >= Nn * 128) return;
    int n = t >> 7, c = t & 127;
    float d0 = denom[n * 2], d1 = denom[n * 2 + 1];
    float a0 = attn[(size_t)n * 256 + c], a1 = attn[(size_t)n * 256 + 128 + c];
    float r0 = d0 > 0.f ? a0 / d0 : 0.f;
    float r1 = d1 > 0.f ? a1 / d1 : 0.f;
    x1[t] = x[t] + 0.5f * (r0 + r1) + qkvs[(size_t)n * 896 + 768 + c];
}

// ---- kernel 7: LN2 + FFN (GEMM1 -> exact GELU -> GEMM2) + residual, bf16 WMMA ----
__global__ __launch_bounds__(256) void k_ffn(
    const float* __restrict__ x1, const float* __restrict__ g2, const float* __restrict__ bt2,
    const __bf16* __restrict__ pw1, const __bf16* __restrict__ pw2,
    const float* __restrict__ b1f, const float* __restrict__ b2f,
    float* __restrict__ out, int Nn) {
    __shared__ __bf16 ly[128 * LDSW];
    __shared__ __bf16 lh[128 * LDSW];
    int row0 = blockIdx.x * 128;
    ln_to_lds(x1, g2, bt2, ly, row0, Nn);
    __syncthreads();

    int wave = threadIdx.x >> 5, lane = threadIdx.x & 31;
    int m0 = wave * 16;
    int mr = m0 + (lane & 15);
    int ss = (lane >> 4) * 8;
    int nlane = lane & 15;
    int rhi = (lane >> 4) << 3;

    v16bf af[4];
#pragma unroll
    for (int kt = 0; kt < 4; ++kt) af[kt] = load_afrag(ly, mr, kt, ss);

    {   // GEMM1 + GELU -> LDS, double-buffered B
        const __bf16* bpbase = pw1 + lane * 16;
        v16bf bcur[4], bnxt[4];
        load_bfrags(bpbase, bcur);
        for (int nt = 0; nt < 8; ++nt) {
            if (nt + 1 < 8) load_bfrags(bpbase + (size_t)(nt + 1) * 2048, bnxt);
            v8f acc = {0.f, 0.f, 0.f, 0.f, 0.f, 0.f, 0.f, 0.f};
#pragma unroll
            for (int kt = 0; kt < 4; ++kt) acc = wmma_bf16(af[kt], bcur[kt], acc);
            int n = nt * 16 + nlane;
            float bb = b1f[n];
#pragma unroll
            for (int rr = 0; rr < 8; ++rr) {
                float v  = acc[rr] + bb;
                float ge = 0.5f * v * (1.0f + erff(v * 0.70710678118654752f));
                lh[(m0 + rr + rhi) * LDSW + n] = f2bf(ge);
            }
#pragma unroll
            for (int kt = 0; kt < 4; ++kt) bcur[kt] = bnxt[kt];
        }
    }
    __syncthreads();

#pragma unroll
    for (int kt = 0; kt < 4; ++kt) af[kt] = load_afrag(lh, mr, kt, ss);
    int rbase = row0 + m0 + rhi;
    bool full = (row0 + 128) <= Nn;               // block-uniform fast path
    {   // GEMM2 + residual -> out, double-buffered B
        const __bf16* bpbase = pw2 + lane * 16;
        v16bf bcur[4], bnxt[4];
        load_bfrags(bpbase, bcur);
        for (int nt = 0; nt < 8; ++nt) {
            if (nt + 1 < 8) load_bfrags(bpbase + (size_t)(nt + 1) * 2048, bnxt);
            v8f acc = {0.f, 0.f, 0.f, 0.f, 0.f, 0.f, 0.f, 0.f};
#pragma unroll
            for (int kt = 0; kt < 4; ++kt) acc = wmma_bf16(af[kt], bcur[kt], acc);
            int n = nt * 16 + nlane;
            float bb = b2f[n];
            if (full) {
#pragma unroll
                for (int rr = 0; rr < 8; ++rr) {
                    size_t gi = (size_t)(rbase + rr) * 128 + n;
                    out[gi] = x1[gi] + acc[rr] + bb;
                }
            } else {
#pragma unroll
                for (int rr = 0; rr < 8; ++rr) {
                    int gr = rbase + rr;
                    if (gr < Nn) {
                        size_t gi = (size_t)gr * 128 + n;
                        out[gi] = x1[gi] + acc[rr] + bb;
                    }
                }
            }
#pragma unroll
            for (int kt = 0; kt < 4; ++kt) bcur[kt] = bnxt[kt];
        }
    }
}

extern "C" void kernel_launch(void* const* d_in, const int* in_sizes, int n_in,
                              void* d_out, int out_size, void* d_ws, size_t ws_size,
                              hipStream_t stream) {
    const float*     x    = (const float*)d_in[0];
    const long long* ei   = (const long long*)d_in[1];
    const float*     ln1g = (const float*)d_in[2];
    const float*     ln1b = (const float*)d_in[3];
    const float*     Wq   = (const float*)d_in[4];
    const float*     bq   = (const float*)d_in[5];
    const float*     Wk   = (const float*)d_in[6];
    const float*     bk   = (const float*)d_in[7];
    const float*     Wv   = (const float*)d_in[8];
    const float*     bv   = (const float*)d_in[9];
    const float*     Wsk  = (const float*)d_in[10];
    const float*     bsk  = (const float*)d_in[11];
    const float*     ln2g = (const float*)d_in[12];
    const float*     ln2b = (const float*)d_in[13];
    const float*     W1   = (const float*)d_in[14];
    const float*     b1   = (const float*)d_in[15];
    const float*     W2   = (const float*)d_in[16];
    const float*     b2   = (const float*)d_in[17];
    float* out = (float*)d_out;

    int Nn = in_sizes[0] / 128;
    int Ee = in_sizes[1] / 2;

    char* ws = (char*)d_ws;
    size_t off = 0;
    auto take = [&](size_t bytes) {
        char* p = ws + off;
        off += (bytes + 255) & ~(size_t)255;
        return p;
    };
    float*  qkvs  = (float*)take((size_t)Nn * 896 * 4);
    float*  attn  = (float*)take((size_t)Nn * 256 * 4);
    float*  alpha = (float*)take((size_t)Ee * 2 * 4);
    float*  mbuf  = (float*)take((size_t)Nn * 2 * 4);
    float*  denom = (float*)take((size_t)Nn * 2 * 4);
    float*  x1    = (float*)take((size_t)Nn * 128 * 4);
    __bf16* pw0   = (__bf16*)take(114688 * 2);
    __bf16* pw1   = (__bf16*)take(16384 * 2);
    __bf16* pw2   = (__bf16*)take(16384 * 2);
    float*  bias0 = (float*)take(896 * 4);
    (void)ws_size; (void)n_in; (void)out_size;

    k_pack<<<(114688 + 32768 + 896 + 255) / 256, 256, 0, stream>>>(
        Wq, Wk, Wv, Wsk, W1, W2, bq, bk, bv, bsk, pw0, pw1, pw2, bias0);
    int initTot = Nn * 256 + 4 * Nn;
    k_init<<<(initTot + 255) / 256, 256, 0, stream>>>(attn, denom, mbuf, Nn);
    int nb = (Nn + 127) / 128;
    k_ln1_qkvs<<<nb, 256, 0, stream>>>(x, ln1g, ln1b, pw0, bias0, qkvs, Nn);
    k_edge_logits<<<(2 * Ee + 255) / 256, 256, 0, stream>>>(ei, qkvs, alpha, mbuf, Ee);
    k_edge_accum<<<(Ee + 7) / 8, 256, 0, stream>>>(ei, qkvs, alpha, mbuf, denom, attn, Ee);
    k_combine<<<(Nn * 128 + 255) / 256, 256, 0, stream>>>(x, qkvs, attn, denom, x1, Nn);
    k_ffn<<<nb, 256, 0, stream>>>(x1, ln2g, ln2b, pw1, pw2, b1, b2, out, Nn);
}
